// BaselineBlockNetMultiGraph_4054449127565
// MI455X (gfx1250) — compile-verified, matrix operand order checked by reading, not run on hardware
//
#include <hip/hip_runtime.h>

typedef __attribute__((ext_vector_type(16))) _Float16 v16h;
typedef __attribute__((ext_vector_type(8)))  _Float16 v8h;
typedef __attribute__((ext_vector_type(8)))  float    v8f;

#define WMMA(a, b, c) \
  __builtin_amdgcn_wmma_f32_16x16x32_f16(false, (a), false, (b), (short)0, (c), false, false)

// ---------------- fragment-packed layout helpers ----------------
// A 16x32 (row-major src [row][k]) and B 32x16 (src B[k][col]) fragments are
// stored packed: 512 halves = [lane 0..31][half 0..15].
// Read mapping (per CDNA5 ISA): lane l -> row/col = l&15,
// half i -> k = (i<8 ? i : i+8) + (l&16 ? 8 : 0).
// pk_pos(rc, klocal) = packed offset of element (row/col rc, k klocal in [0,32)).
__device__ __forceinline__ int pk_pos(int rc, int klocal) {
  int l = (rc & 15) + ((klocal & 8) ? 16 : 0);
  int bk = klocal & ~8;                 // {0..7, 16..23}
  int ip = (bk < 8) ? bk : bk - 8;      // {0..7, 8..15}
  return l * 16 + ip;
}
__device__ __forceinline__ v16h frag_ld(const _Float16* p) {
  return *(const v16h*)(p + (threadIdx.x & 31) * 16);   // 32B/lane -> 2x b128
}
// C/D 16x16 f32: lane l -> col = l&15; reg r -> row = r + (l&16 ? 8 : 0)
__device__ __forceinline__ void store_c_f32(float* p, int ld, v8f c) {
  const int l = threadIdx.x & 31;
  const int col = l & 15;
  const int moff = (l & 16) ? 8 : 0;
#pragma unroll
  for (int r = 0; r < 8; ++r) p[(size_t)(r + moff) * ld + col] = c[r];
}
__device__ __forceinline__ float fast_rcp(float x) { return __builtin_amdgcn_rcpf(x); }

// ---------------- GRU: 1 wave = 16 series; packed h + packed whh ----------------
__global__ void __launch_bounds__(128) k_gru(const float* __restrict__ x,
                                             const float* __restrict__ wih,
                                             const float* __restrict__ whh,
                                             const float* __restrict__ bih,
                                             const float* __restrict__ bhh,
                                             float* __restrict__ hT) {
  __shared__ alignas(32) _Float16 whh_pk[24 * 512];   // frag = gateTile*2 + kk
  __shared__ alignas(32) _Float16 h_pk[4][2 * 512];   // per-wave packed A frags
  const int tid = threadIdx.x, wv = tid >> 5, lane = tid & 31;
  const int colL = lane & 15, moff = (lane & 16) ? 8 : 0;

  // stage whh as packed B (source-linear, coalesced): B[k][g] = whh[g][k]
  for (int s = tid; s < 192 * 64; s += 128) {
    int g = s >> 6, k = s & 63;
    whh_pk[((g >> 4) * 2 + (k >> 5)) * 512 + pk_pos(g, k & 31)] = (_Float16)whh[s];
  }
  _Float16* hp = h_pk[wv];
  for (int i = lane; i < 1024; i += 32) hp[i] = (_Float16)0.0f;

  float wr[4], wz[4], wn[4], bir[4], biz[4], binn[4], bhr[4], bhz[4], bhn[4];
#pragma unroll
  for (int j = 0; j < 4; ++j) {
    int cj = j * 16 + colL;
    wr[j] = wih[cj];   wz[j] = wih[64 + cj];  wn[j] = wih[128 + cj];
    bir[j] = bih[cj];  biz[j] = bih[64 + cj]; binn[j] = bih[128 + cj];
    bhr[j] = bhh[cj];  bhz[j] = bhh[64 + cj]; bhn[j] = bhh[128 + cj];
  }
  float hprev[4][8];
#pragma unroll
  for (int j = 0; j < 4; ++j)
#pragma unroll
    for (int r = 0; r < 8; ++r) hprev[j][r] = 0.0f;

  const int s0 = (blockIdx.x * 4 + wv) * 16;
  __syncthreads();

  for (int t = 0; t < 32; ++t) {
    float xv[8];
#pragma unroll
    for (int r = 0; r < 8; ++r) {
      int ser = s0 + r + moff;
      xv[r] = x[((ser >> 6) * 32 + t) * 64 + (ser & 63)];
    }
    v16h a0 = frag_ld(hp);
    v16h a1 = frag_ld(hp + 512);
#pragma unroll
    for (int j = 0; j < 4; ++j) {
      v8f aR = {}, aZ = {}, aN = {};
      aR = WMMA(a0, frag_ld(&whh_pk[(j * 2 + 0) * 512]), aR);
      aR = WMMA(a1, frag_ld(&whh_pk[(j * 2 + 1) * 512]), aR);
      aZ = WMMA(a0, frag_ld(&whh_pk[((4 + j) * 2 + 0) * 512]), aZ);
      aZ = WMMA(a1, frag_ld(&whh_pk[((4 + j) * 2 + 1) * 512]), aZ);
      aN = WMMA(a0, frag_ld(&whh_pk[((8 + j) * 2 + 0) * 512]), aN);
      aN = WMMA(a1, frag_ld(&whh_pk[((8 + j) * 2 + 1) * 512]), aN);
#pragma unroll
      for (int r = 0; r < 8; ++r) {
        float ghr = aR[r] + bhr[j];
        float ghz = aZ[r] + bhz[j];
        float ghn = aN[r] + bhn[j];
        float gir = xv[r] * wr[j] + bir[j];
        float giz = xv[r] * wz[j] + biz[j];
        float gin = xv[r] * wn[j] + binn[j];
        float rg = fast_rcp(1.0f + __expf(-(gir + ghr)));
        float zg = fast_rcp(1.0f + __expf(-(giz + ghz)));
        float u = gin + rg * ghn;
        float e2 = __expf(2.0f * u);
        float ng = 1.0f - 2.0f * fast_rcp(e2 + 1.0f);   // tanh(u)
        hprev[j][r] = (1.0f - zg) * ng + zg * hprev[j][r];
      }
    }
    // write h' back in packed-A layout (wave-private; DS in-order per wave)
#pragma unroll
    for (int j = 0; j < 4; ++j)
#pragma unroll
      for (int r = 0; r < 8; ++r) {
        int c = j * 16 + colL;
        hp[(c >> 5) * 512 + pk_pos(r + moff, c & 31)] = (_Float16)hprev[j][r];
      }
  }
  // decode packed h -> hT f32 row-major
  for (int i = lane; i < 1024; i += 32) {
    int kk = i >> 9, l = (i >> 4) & 31, ip = i & 15;
    int row = l & 15;
    int k = kk * 32 + ((ip < 8) ? ip : ip + 8) + ((l & 16) ? 8 : 0);
    hT[(size_t)(s0 + row) * 64 + k] = (float)hp[i];
  }
}

// ---------------- attention + symmetric GCN norm (tiny, scalar) ----------------
__global__ void __launch_bounds__(256) k_attn(const float* __restrict__ hT,
                                              const float* __restrict__ wq_w,
                                              const float* __restrict__ wq_b,
                                              const float* __restrict__ wk_w,
                                              const float* __restrict__ wk_b,
                                              float* __restrict__ What) {
  __shared__ float hs[64 * 64];
  __shared__ float Qs[64 * 32];
  __shared__ float Ks[64 * 32];
  __shared__ float Ss[64 * 64];
  __shared__ float dinv[64];
  const int b = blockIdx.x;
  const int tid = threadIdx.x;
  for (int i = tid; i < 4096; i += 256) hs[i] = hT[(size_t)b * 4096 + i];
  __syncthreads();
  for (int i = tid; i < 64 * 32; i += 256) {
    int n = i >> 5, q = i & 31;
    float sq = wq_b[q], sk = wk_b[q];
    for (int k = 0; k < 64; ++k) {
      float h = hs[n * 64 + k];
      sq += h * wq_w[q * 64 + k];
      sk += h * wk_w[q * 64 + k];
    }
    Qs[i] = sq; Ks[i] = sk;
  }
  __syncthreads();
  const float scale = 0.17677669529663687f;  // 1/sqrt(32)
  for (int i = tid; i < 4096; i += 256) {
    int n = i >> 6, m = i & 63;
    float s = 0.f;
    for (int q = 0; q < 32; ++q) s += Qs[n * 32 + q] * Ks[m * 32 + q];
    Ss[i] = s * scale;
  }
  __syncthreads();
  if (tid < 64) {
    float mx = -1e30f;
    for (int m = 0; m < 64; ++m) mx = fmaxf(mx, Ss[tid * 64 + m]);
    float sum = 0.f;
    for (int m = 0; m < 64; ++m) { float e = __expf(Ss[tid * 64 + m] - mx); Ss[tid * 64 + m] = e; sum += e; }
    float inv = fast_rcp(sum);
    for (int m = 0; m < 64; ++m) Ss[tid * 64 + m] *= inv;
  }
  __syncthreads();
  if (tid < 64) {
    float d = 0.f;
    for (int n = 0; n < 64; ++n) d += Ss[n * 64 + tid];
    dinv[tid] = (d > 0.f) ? rsqrtf(d) : 0.0f;
  }
  __syncthreads();
  for (int i = tid; i < 4096; i += 256) {
    int n = i >> 6, m = i & 63;
    What[(size_t)b * 4096 + i] = dinv[n] * Ss[i] * dinv[m];
  }
}

// ---------------- x -> inter dim, writing fragment-packed feats ----------------
// feats tile (t,b): 8 frags (mi,kk) of A[n][c]; element decode below.
__global__ void k_x2i(const float* __restrict__ x, const float* __restrict__ w,
                      const float* __restrict__ bw, _Float16* __restrict__ feats) {
  int idx = blockIdx.x * 256 + threadIdx.x;   // 4M total, grid exact
  int tile = idx >> 12;                        // t*32 + b
  int t = tile >> 5, b = tile & 31;
  int f = (idx >> 9) & 7, l = (idx >> 4) & 31, ip = idx & 15;
  int n = (f >> 1) * 16 + (l & 15);
  int c = (f & 1) * 32 + ((ip < 8) ? ip : ip + 8) + ((l & 16) ? 8 : 0);
  float xv = x[(b * 32 + t) * 64 + n];
  feats[idx] = (_Float16)(xv * w[c] + bw[c]);
}

// ---------------- GCN block: agg = What_b^T @ (X @ theta_t) + gb (all packed) ----------------
__global__ void __launch_bounds__(128) k_gcn(const _Float16* __restrict__ fin,
                                             const float* __restrict__ What,
                                             const float* __restrict__ gw,
                                             const float* __restrict__ gb,
                                             _Float16* __restrict__ agg) {
  __shared__ alignas(32) _Float16 Wg[8 * 512];   // packed B frags (ni,kk)
  __shared__ alignas(32) _Float16 Wh[8 * 512];   // packed A^T frags (mi,kk)
  __shared__ alignas(32) _Float16 Ts[8 * 512];   // packed B frags of tmp
  const int tid = threadIdx.x, wv = tid >> 5, lane = tid & 31;
  const int colL = lane & 15, moff = (lane & 16) ? 8 : 0;
  const int t = blockIdx.x >> 5, b = blockIdx.x & 31;

  // Wg: B[k=c][n=d] = gw[t][c][d]; source-linear s = c*64+d (coalesced)
  for (int s = tid; s < 4096; s += 128) {
    int c = s >> 6, d = s & 63;
    Wg[((d >> 4) * 2 + (c >> 5)) * 512 + pk_pos(d, c & 31)] = (_Float16)gw[(size_t)t * 4096 + s];
  }
  // Wh: A[j][i] = What[b][i][j]; source-linear s = i*64+j (coalesced)
  for (int s = tid; s < 4096; s += 128) {
    int i = s >> 6, j = s & 63;
    Wh[((j >> 4) * 2 + (i >> 5)) * 512 + pk_pos(j, i & 31)] = (_Float16)What[(size_t)b * 4096 + s];
  }
  __syncthreads();

  const _Float16* xin = fin + (size_t)(t * 32 + b) * 4096;
  v16h xa0 = frag_ld(xin + (wv * 2 + 0) * 512);   // global b128 loads
  v16h xa1 = frag_ld(xin + (wv * 2 + 1) * 512);
#pragma unroll
  for (int ni = 0; ni < 4; ++ni) {
    v8f acc = {};
    acc = WMMA(xa0, frag_ld(&Wg[(ni * 2 + 0) * 512]), acc);
    acc = WMMA(xa1, frag_ld(&Wg[(ni * 2 + 1) * 512]), acc);
    // store C (row = n = wv*16+r+moff, col = d = ni*16+colL) into packed-B Ts[k=n][d]
#pragma unroll
    for (int r = 0; r < 8; ++r) {
      int kg = wv * 16 + r + moff;
      Ts[(ni * 2 + (kg >> 5)) * 512 + pk_pos(colL, kg & 31)] = (_Float16)acc[r];
    }
  }
  __syncthreads();
  v16h wa0 = frag_ld(&Wh[(wv * 2 + 0) * 512]);
  v16h wa1 = frag_ld(&Wh[(wv * 2 + 1) * 512]);
  _Float16* aout = agg + (size_t)(t * 32 + b) * 4096;
#pragma unroll
  for (int ni = 0; ni < 4; ++ni) {
    v8f acc = {};
    acc = WMMA(wa0, frag_ld(&Ts[(ni * 2 + 0) * 512]), acc);
    acc = WMMA(wa1, frag_ld(&Ts[(ni * 2 + 1) * 512]), acc);
    float gbv = gb[t * 64 + ni * 16 + colL];
    // store into packed-A global: element (row = n, k = channel c)
#pragma unroll
    for (int r = 0; r < 8; ++r) {
      int cg = ni * 16 + colL;
      int row = wv * 16 + r + moff;
      aout[((row >> 4) * 2 + (cg >> 5)) * 512 + pk_pos(row, cg & 31)] = (_Float16)(acc[r] + gbv);
    }
  }
}

// ---------------- temporal conv + LeakyReLU (packed in/out) ----------------
__global__ void __launch_bounds__(128) k_conv(const _Float16* __restrict__ agg,
                                              const float* __restrict__ cw,
                                              const float* __restrict__ cb,
                                              _Float16* __restrict__ fout, int ksz) {
  __shared__ alignas(32) _Float16 Bs[8 * 512];
  const int tid = threadIdx.x, wv = tid >> 5, lane = tid & 31;
  const int colL = lane & 15, moff = (lane & 16) ? 8 : 0;
  const int t = blockIdx.x >> 5, b = blockIdx.x & 31;
  const int p = ksz >> 1;
  v8f acc[4] = {};
  for (int dt = 0; dt < ksz; ++dt) {
    int tin = t + dt - p;              // uniform across block
    if (tin < 0 || tin >= 32) continue;
    __syncthreads();
    // Bs: B[k=c][n=d] = cw[d][c][dt]; s = d*64+c keeps global stride = ksz
    for (int s = tid; s < 4096; s += 128) {
      int d = s >> 6, c = s & 63;
      Bs[((d >> 4) * 2 + (c >> 5)) * 512 + pk_pos(d, c & 31)] =
          (_Float16)cw[(size_t)(d * 64 + c) * ksz + dt];
    }
    __syncthreads();
    const _Float16* ain = agg + (size_t)(tin * 32 + b) * 4096;
    v16h a0 = frag_ld(ain + (wv * 2 + 0) * 512);
    v16h a1 = frag_ld(ain + (wv * 2 + 1) * 512);
#pragma unroll
    for (int ni = 0; ni < 4; ++ni) {
      acc[ni] = WMMA(a0, frag_ld(&Bs[(ni * 2 + 0) * 512]), acc[ni]);
      acc[ni] = WMMA(a1, frag_ld(&Bs[(ni * 2 + 1) * 512]), acc[ni]);
    }
  }
  _Float16* aout = fout + (size_t)(t * 32 + b) * 4096;
#pragma unroll
  for (int ni = 0; ni < 4; ++ni) {
    float cbv = cb[ni * 16 + colL];
#pragma unroll
    for (int r = 0; r < 8; ++r) {
      float v = acc[ni][r] + cbv;
      v = (v >= 0.f) ? v : 0.01f * v;
      int cg = ni * 16 + colL;
      int row = wv * 16 + r + moff;
      aout[((row >> 4) * 2 + (cg >> 5)) * 512 + pk_pos(row, cg & 31)] = (_Float16)v;
    }
  }
}

// ---------------- packed feats -> flat[b][f] row-major (f = n*2048 + c*32 + t) ----------------
__global__ void __launch_bounds__(128) k_transp(const _Float16* __restrict__ fpk,
                                                _Float16* __restrict__ flat) {
  const int b = blockIdx.x >> 6, n = blockIdx.x & 63;
  const int fbase = (n >> 4) * 2;
  for (int w = threadIdx.x; w < 256; w += 128) {
    int c = w >> 2, t0 = (w & 3) * 8;
    int off = (fbase + (c >> 5)) * 512 + pk_pos(n, c & 31);
    v8h o;
#pragma unroll
    for (int u = 0; u < 8; ++u) {
      int t = t0 + u;
      o[u] = fpk[(size_t)(t * 32 + b) * 4096 + off];
    }
    *(v8h*)(flat + (size_t)b * 131072 + n * 2048 + c * 32 + t0) = o;   // 16B store
  }
}

// ---------------- final linear: LDS-free, fragments straight from global ----------------
__global__ void __launch_bounds__(32) k_flat(const _Float16* __restrict__ flat,
                                             const float* __restrict__ lw,
                                             float* __restrict__ partials) {
  const int ntile = blockIdx.x % 48, kz = blockIdx.x / 48;
  const int lane = threadIdx.x & 31;
  const int o0 = ntile * 16;
  const int r15 = lane & 15;
  const int koff = (lane & 16) ? 8 : 0;
  v8f acc0 = {}, acc1 = {};
  const size_t fbase = (size_t)kz * 128 * 32;
  for (int q = 0; q < 128; ++q) {
    const size_t f0 = fbase + (size_t)q * 32;
    const _Float16* pa0 = flat + (size_t)r15 * 131072 + f0 + koff;
    const _Float16* pa1 = pa0 + (size_t)16 * 131072;
    const float*    pw  = lw + (size_t)(o0 + r15) * 131072 + f0 + koff;
    v16h a0, a1, bf;
#pragma unroll
    for (int i = 0; i < 8; ++i) {
      a0[i] = pa0[i];      a0[8 + i] = pa0[16 + i];
      a1[i] = pa1[i];      a1[8 + i] = pa1[16 + i];
      bf[i] = (_Float16)pw[i];
      bf[8 + i] = (_Float16)pw[16 + i];
    }
    acc0 = WMMA(a0, bf, acc0);
    acc1 = WMMA(a1, bf, acc1);
  }
  float* dst = partials + (size_t)kz * 24576 + o0;   // [kz][b][768]
  store_c_f32(dst, 768, acc0);
  store_c_f32(dst + (size_t)16 * 768, 768, acc1);
}

// ---------------- deterministic partial reduction + bias ----------------
__global__ void k_reduce(const float* __restrict__ partials, const float* __restrict__ lb,
                         float* __restrict__ out) {
  int idx = blockIdx.x * 256 + threadIdx.x;  // 24576 = 96*256, grid exact
  int o = idx % 768;
  float s = lb[o];
  for (int kz = 0; kz < 32; ++kz) s += partials[(size_t)kz * 24576 + idx];
  out[idx] = s;
}

extern "C" void kernel_launch(void* const* d_in, const int* in_sizes, int n_in,
                              void* d_out, int out_size, void* d_ws, size_t ws_size,
                              hipStream_t stream) {
  (void)in_sizes; (void)n_in; (void)out_size; (void)ws_size;
  const float* x        = (const float*)d_in[0];
  const float* w_x2i    = (const float*)d_in[1];
  const float* b_x2i    = (const float*)d_in[2];
  const float* gru_wih  = (const float*)d_in[3];
  const float* gru_whh  = (const float*)d_in[4];
  const float* gru_bih  = (const float*)d_in[5];
  const float* gru_bhh  = (const float*)d_in[6];
  const float* wq_w     = (const float*)d_in[7];
  const float* wq_b     = (const float*)d_in[8];
  const float* wk_w     = (const float*)d_in[9];
  const float* wk_b     = (const float*)d_in[10];
  const float* gcn_w[3]  = {(const float*)d_in[11], (const float*)d_in[15], (const float*)d_in[19]};
  const float* gcn_b[3]  = {(const float*)d_in[12], (const float*)d_in[16], (const float*)d_in[20]};
  const float* conv_w[3] = {(const float*)d_in[13], (const float*)d_in[17], (const float*)d_in[21]};
  const float* conv_b[3] = {(const float*)d_in[14], (const float*)d_in[18], (const float*)d_in[22]};
  const float* lout_w   = (const float*)d_in[23];
  const float* lout_b   = (const float*)d_in[24];
  float* out = (float*)d_out;

  char* ws = (char*)d_ws;
  size_t off = 0;
  float*     hT       = (float*)(ws + off);     off += (size_t)2048 * 64 * 4;   // 512 KB
  float*     What     = (float*)(ws + off);     off += (size_t)32 * 4096 * 4;   // 512 KB
  _Float16*  featsA   = (_Float16*)(ws + off);  off += (size_t)4194304 * 2;     // 8 MB
  _Float16*  featsB   = (_Float16*)(ws + off);  off += (size_t)4194304 * 2;     // 8 MB
  _Float16*  aggB     = (_Float16*)(ws + off);  off += (size_t)4194304 * 2;     // 8 MB
  float*     partials = (float*)(ws + off);     off += (size_t)32 * 24576 * 4;  // 3 MB

  k_gru<<<dim3(32), dim3(128), 0, stream>>>(x, gru_wih, gru_whh, gru_bih, gru_bhh, hT);
  k_attn<<<dim3(32), dim3(256), 0, stream>>>(hT, wq_w, wq_b, wk_w, wk_b, What);
  k_x2i<<<dim3(16384), dim3(256), 0, stream>>>(x, w_x2i, b_x2i, featsA);

  _Float16* fin = featsA;
  _Float16* fout = featsB;
  const int ks[3] = {3, 5, 7};
  for (int i = 0; i < 3; ++i) {
    k_gcn<<<dim3(1024), dim3(128), 0, stream>>>(fin, What, gcn_w[i], gcn_b[i], aggB);
    k_conv<<<dim3(1024), dim3(128), 0, stream>>>(aggB, conv_w[i], conv_b[i], fout, ks[i]);
    _Float16* tmp = fin; fin = fout; fout = tmp;
  }
  // fin == featsB holds final packed feats; featsA is dead -> reuse as flat buffer
  _Float16* flatT = featsA;
  k_transp<<<dim3(2048), dim3(128), 0, stream>>>(fin, flatT);
  k_flat<<<dim3(48 * 32), dim3(32), 0, stream>>>(flatT, lout_w, partials);
  k_reduce<<<dim3(96), dim3(256), 0, stream>>>(partials, lout_b, out);
}
